// MiniBatchDiscrimination_52140902973873
// MI455X (gfx1250) — compile-verified, matrix operand order checked by reading, not run on hardware
//
#include <hip/hip_runtime.h>
#include <hip/hip_bf16.h>

#define B_SZ 512
#define F_SZ 512
#define NK 50
#define DK 16
#define N_SZ (NK * DK)      // 800
#define OUTW (F_SZ + NK)    // 562

typedef __attribute__((ext_vector_type(2)))  float    v2f;
typedef __attribute__((ext_vector_type(8)))  float    v8f;
typedef __attribute__((ext_vector_type(16))) _Float16 v16h;

#if __has_builtin(__builtin_amdgcn_wmma_f32_16x16x4_f32)
#define USE_F32_WMMA 1
#else
#define USE_F32_WMMA 0
#endif

// ---------------------------------------------------------------------------
// Kernel 1: act[512][800] = X[512][512] @ W[512][800] via WMMA.
// One 16x16 output tile per wave32. 32 (M) x 50 (N) = 1600 tiles
// = 200 blocks * 8 waves exactly -> no divergence, EXEC all-ones for WMMA.
// ---------------------------------------------------------------------------
__global__ __launch_bounds__(256) void mbd_gemm_wmma(const float* __restrict__ X,
                                                     const float* __restrict__ W,
                                                     float* __restrict__ act) {
  const int lane = threadIdx.x & 31;
  const int wave = threadIdx.x >> 5;
  const int tile = blockIdx.x * 8 + wave;   // 0..1599
  const int tileM = tile & 31;              // 512/16 = 32 tiles along M
  const int tileN = tile >> 5;              // 800/16 = 50 tiles along N
  const int m0 = tileM * 16;
  const int n0 = tileN * 16;
  const int half = lane >> 4;               // 0: lanes 0-15, 1: lanes 16-31
  const int l15  = lane & 15;

  v8f c = {};
  const float* Arow = X + (size_t)(m0 + l15) * F_SZ;  // row m of A
  const float* Bcol = W + (n0 + l15);                 // column n of B

#if USE_F32_WMMA
  // ISA 32-bit A 16x4 layout: VGPR0 = K{0 | 2}, VGPR1 = K{1 | 3} (lo|hi half-wave).
  for (int k = 0; k < F_SZ; k += 4) {
    const int ka = k + 2 * half;
    v2f a, b;
    a.x = Arow[ka];
    a.y = Arow[ka + 1];
    b.x = Bcol[(size_t)ka * N_SZ];
    b.y = Bcol[(size_t)(ka + 1) * N_SZ];
    c = __builtin_amdgcn_wmma_f32_16x16x4_f32(false, a, false, b,
                                              (short)0, c, false, false);
  }
#else
  // Fallback: probe-confirmed f16 WMMA 16x16x32 with on-the-fly f32->f16 cvt.
  // 16-bit A 16x32 layout: VGPR j in 0..3 -> K = half*8 + 2j + {0,1};
  //                        VGPR j in 4..7 -> K = 16 + half*8 + 2(j-4) + {0,1}.
  for (int k = 0; k < F_SZ; k += 32) {
    v16h a, b;
#pragma unroll
    for (int s = 0; s < 16; ++s) {
      const int vg  = s >> 1;
      const int pos = s & 1;
      const int ko  = ((vg >= 4) ? 16 : 0) + (half ? 8 : 0) + 2 * (vg & 3) + pos;
      a[s] = (_Float16)Arow[k + ko];
      b[s] = (_Float16)Bcol[(size_t)(k + ko) * N_SZ];
    }
    c = __builtin_amdgcn_wmma_f32_16x16x32_f16(false, a, false, b,
                                               (short)0, c, false, false);
  }
#endif

  // C/D layout: lane half selects M block (0-7 vs 8-15), VGPR r = row within block.
#pragma unroll
  for (int r = 0; r < 8; ++r) {
    act[(size_t)(m0 + 8 * half + r) * N_SZ + (n0 + l15)] = c[r];
  }
}

// ---------------------------------------------------------------------------
// Kernel 2: features[b][k] = sum_b2 exp(-sum_d |act[b,k,d] - act[b2,k,d]|)
// One block per k (50 blocks, 512 threads). Stage act[:,k,:] (32 KB) in LDS;
// inner loop reads are wave-uniform -> LDS broadcast, no bank conflicts.
// ---------------------------------------------------------------------------
__global__ __launch_bounds__(512) void mbd_feat(const float* __restrict__ act,
                                                float* __restrict__ out) {
  const int k = blockIdx.x;    // 0..49
  const int b = threadIdx.x;   // 0..511

  __shared__ float sa[B_SZ][DK];   // 32 KB of the 320 KB WGP LDS

  for (int i = threadIdx.x; i < B_SZ * DK; i += B_SZ) {
    const int row = i >> 4;
    const int d   = i & 15;
    sa[row][d] = act[(size_t)row * N_SZ + k * DK + d];
  }
  __syncthreads();

  float mine[DK];
#pragma unroll
  for (int d = 0; d < DK; ++d) mine[d] = sa[b][d];

  float feat = 0.0f;
  for (int b2 = 0; b2 < B_SZ; ++b2) {
    float l1 = 0.0f;
#pragma unroll
    for (int d = 0; d < DK; ++d) {
      l1 += fabsf(mine[d] - sa[b2][d]);   // v_sub + v_add with |src| modifier
    }
    feat += __expf(-l1);                   // v_exp_f32 (self term -> +1)
  }

  out[(size_t)b * OUTW + F_SZ + k] = feat;
}

// ---------------------------------------------------------------------------
// Kernel 3: out[:, 0:512] = x
// ---------------------------------------------------------------------------
__global__ __launch_bounds__(256) void mbd_copy(const float* __restrict__ X,
                                                float* __restrict__ out) {
  const int i = blockIdx.x * blockDim.x + threadIdx.x;  // 0 .. 512*512-1
  const int b = i >> 9;
  const int f = i & 511;
  out[(size_t)b * OUTW + f] = X[i];
}

extern "C" void kernel_launch(void* const* d_in, const int* in_sizes, int n_in,
                              void* d_out, int out_size, void* d_ws, size_t ws_size,
                              hipStream_t stream) {
  const float* x = (const float*)d_in[0];   // (512, 512) f32
  const float* W = (const float*)d_in[1];   // (512, 800) f32
  float* out = (float*)d_out;               // (512, 562) f32
  float* act = (float*)d_ws;                // scratch: 512*800 f32 = 1.64 MB

  mbd_gemm_wmma<<<200, 256, 0, stream>>>(x, W, act);
  mbd_feat<<<NK, 512, 0, stream>>>(act, out);
  mbd_copy<<<(B_SZ * F_SZ) / 256, 256, 0, stream>>>(x, out);
}
